// GRU_73701638799447
// MI455X (gfx1250) — matrix operand order repaired and checked by reading, not compiled, not run on hardware
//
#include <hip/hip_runtime.h>
#include <math.h>

// ---- sizes ----
#define S_ 128
#define B_ 64
#define E_ 512
#define H_ 1024
#define V_ 10000
#define SB_ (S_*B_)
#define SBH_ ((size_t)SB_*(size_t)H_)
#define SCAN_BLOCKS 16

typedef __bf16 bf16;
typedef __attribute__((ext_vector_type(8)))  __bf16 v8bf;
typedef __attribute__((ext_vector_type(16))) __bf16 v16bf;
typedef __attribute__((ext_vector_type(8)))  float  v8f;

// ---------------- WMMA fragment loaders (bf16, 16x16x32) ----------------
// A is [M,K] row-major, pre-offset to tile row m0.
// ISA 16-bit A layout: lane<16 row M=lane holds K={h*8..+7, h*8+16..+23}, h=lane>>4
__device__ __forceinline__ v16bf load_a_frag(const bf16* A, int lda, int k0, int lane) {
  int row = lane & 15, half = lane >> 4;
  const bf16* p = A + (size_t)row * lda + k0 + half * 8;
  v8bf lo = *(const v8bf*)p;          // K = base .. base+7
  v8bf hi = *(const v8bf*)(p + 16);   // K = base+16 .. base+23
  return __builtin_shufflevector(lo, hi, 0,1,2,3,4,5,6,7,8,9,10,11,12,13,14,15);
}
// W is [N,K] row-major (acts as B^T), pre-offset to tile row n0.
// ISA 16-bit B layout: lane holds column N=lane&15, K={h*8..+7, h*8+16..+23}, h=lane>>4
__device__ __forceinline__ v16bf load_b_frag(const bf16* W, int ldb, int k0, int lane) {
  int col = lane & 15, half = lane >> 4;
  const bf16* p = W + (size_t)col * ldb + k0 + half * 8;
  v8bf lo = *(const v8bf*)p;
  v8bf hi = *(const v8bf*)(p + 16);
  return __builtin_shufflevector(lo, hi, 0,1,2,3,4,5,6,7,8,9,10,11,12,13,14,15);
}

// One wave computes a 16x(NT*16) strip; A fragment reused NT times.
// All NT B-tile loads are issued before the WMMA chain -> loads overlap compute.
template<int NT>
__device__ __forceinline__ void mm_strip(v8f* acc, const bf16* A, int lda,
                                         const bf16* W, int ldb, int K, int lane) {
  for (int k0 = 0; k0 < K; k0 += 32) {
    v16bf a = load_a_frag(A, lda, k0, lane);
    v16bf b[NT];
#pragma unroll
    for (int t = 0; t < NT; ++t)
      b[t] = load_b_frag(W + (size_t)(t * 16) * ldb, ldb, k0, lane);
#pragma unroll
    for (int t = 0; t < NT; ++t)
      acc[t] = __builtin_amdgcn_wmma_f32_16x16x32_bf16(
          false, a, false, b[t], (short)0, acc[t], false, false);
  }
}

__device__ __forceinline__ int global_wave() {
  return __builtin_amdgcn_readfirstlane(
      (int)((blockIdx.x * blockDim.x + threadIdx.x) >> 5));
}

// ---------------- generic GEMM  C[M,N] = A[M,K] * W[N,K]^T (+bias) ----------------
template<int NT>
__global__ void k_gemm(const bf16* __restrict__ A, int lda,
                       const bf16* __restrict__ W, int ldb,
                       const float* __restrict__ bias,
                       float* __restrict__ C, int ldc,
                       int M, int N, int K) {
  int lane = threadIdx.x & 31;
  int wave = global_wave();
  int strips = N / (NT * 16);
  int mT = wave / strips, nS = wave % strips;
  if (mT * 16 >= M) return;                 // scalar branch, EXEC untouched
  int m0 = mT * 16, n0 = nS * (NT * 16);
  v8f acc[NT] = {};
  mm_strip<NT>(acc, A + (size_t)m0 * lda, lda, W + (size_t)n0 * ldb, ldb, K, lane);
  int col = lane & 15, rb = (lane >> 4) * 8;
#pragma unroll
  for (int t = 0; t < NT; ++t) {
    int n = n0 + t * 16 + col;
    float bv = bias ? bias[n] : 0.f;
#pragma unroll
    for (int i = 0; i < 8; ++i)
      C[(size_t)(m0 + rb + i) * ldc + n] = acc[t][i] + bv;
  }
}

// ---------------- persistent-scan phase bodies ----------------
// r/z gates: pre = A1*B1^T (+A2*B2^T) (+xp) + bias over N=2H.
// r -> rh_out = sigmoid(.)*hprev (bf16); z -> z_out (f32).  128 wave-tasks (NT=4).
__device__ __forceinline__ void phase_rz(int wave, int lane,
    const bf16* A1, const bf16* B1, const bf16* A2, const bf16* B2,
    const float* xp, size_t xp_gstride, const float* bias,
    const float* hprev, bf16* rh_out, float* z_out) {
  const int strips = (2 * H_) >> 6;          // 32
  int mT = wave / strips, nS = wave % strips;
  if (mT * 16 >= B_) return;
  int m0 = mT * 16, n0 = nS * 64;
  v8f acc[4] = {};
  mm_strip<4>(acc, A1 + (size_t)m0 * H_, H_, B1 + (size_t)n0 * H_, H_, H_, lane);
  if (A2)
    mm_strip<4>(acc, A2 + (size_t)m0 * H_, H_, B2 + (size_t)n0 * H_, H_, H_, lane);
  int col = lane & 15, rb = (lane >> 4) * 8;
#pragma unroll
  for (int t = 0; t < 4; ++t) {
    int gt = (n0 + t * 16) >> 10;            // scalar: tiles never straddle H
    int n = n0 + t * 16 + col;
    int o = n & (H_ - 1);
#pragma unroll
    for (int i = 0; i < 8; ++i) {
      int m = m0 + rb + i;
      float v = acc[t][i] + bias[n];
      if (xp) v += xp[(size_t)gt * xp_gstride + (size_t)m * H_ + o];
      float sg = 1.f / (1.f + __expf(-v));
      size_t idx = (size_t)m * H_ + o;
      if (gt == 0) rh_out[idx] = (bf16)(sg * hprev[idx]);
      else         z_out[idx]  = sg;
    }
  }
}

// h~ + state update over N=H: hn = (1-z)*h + z*tanh(pre).  128 wave-tasks (NT=2).
__device__ __forceinline__ void phase_h(int wave, int lane,
    const bf16* A1, const bf16* B1, const bf16* A2, const bf16* B2,
    const float* xp, const float* bias_h, const float* zbuf,
    float* h, bf16* hb, bf16* hs_out) {
  const int strips = H_ >> 5;                // 32 strips of 32 cols
  int mT = wave / strips, nS = wave % strips;
  if (mT * 16 >= B_) return;
  int m0 = mT * 16, n0 = nS * 32;
  v8f acc[2] = {};
  mm_strip<2>(acc, A1 + (size_t)m0 * H_, H_, B1 + (size_t)n0 * H_, H_, H_, lane);
  if (A2)
    mm_strip<2>(acc, A2 + (size_t)m0 * H_, H_, B2 + (size_t)n0 * H_, H_, H_, lane);
  int col = lane & 15, rb = (lane >> 4) * 8;
#pragma unroll
  for (int t = 0; t < 2; ++t) {
    int n = n0 + t * 16 + col;
#pragma unroll
    for (int i = 0; i < 8; ++i) {
      int m = m0 + rb + i;
      float v = acc[t][i] + bias_h[n];
      size_t idx = (size_t)m * H_ + n;
      if (xp) v += xp[idx];
      float ht = tanhf(v);
      float z  = zbuf[idx];
      float hn = (1.f - z) * h[idx] + z * ht;
      h[idx]  = hn;
      hb[idx] = (bf16)hn;
      if (hs_out) hs_out[idx] = (bf16)hn;
    }
  }
}

// ---------------- grid-wide barrier (sense counter, agent scope) ----------------
__device__ __forceinline__ void grid_barrier(int* ctr) {
  __syncthreads();
  if (threadIdx.x == 0) {
    __threadfence();                          // release this block's writes
    int gen = __hip_atomic_load(&ctr[1], __ATOMIC_RELAXED, __HIP_MEMORY_SCOPE_AGENT);
    int arrived = __hip_atomic_fetch_add(&ctr[0], 1, __ATOMIC_ACQ_REL,
                                         __HIP_MEMORY_SCOPE_AGENT);
    if (arrived == SCAN_BLOCKS - 1) {
      __hip_atomic_store(&ctr[0], 0, __ATOMIC_RELAXED, __HIP_MEMORY_SCOPE_AGENT);
      __hip_atomic_store(&ctr[1], gen + 1, __ATOMIC_RELEASE, __HIP_MEMORY_SCOPE_AGENT);
    } else {
      while (__hip_atomic_load(&ctr[1], __ATOMIC_ACQUIRE,
                               __HIP_MEMORY_SCOPE_AGENT) == gen)
        __builtin_amdgcn_s_sleep(2);
    }
    __threadfence();                          // acquire for subsequent reads
  }
  __syncthreads();
}

// ---------------- persistent scan kernel: all 128 steps in one launch ----------------
__global__ void __launch_bounds__(256)
k_scan(const bf16* __restrict__ Uh0b, const bf16* __restrict__ Wx1b,
       const bf16* __restrict__ Uh1b,
       const float* __restrict__ b0, const float* __restrict__ b1,
       const float* __restrict__ xa0,
       float* h0, float* h1, bf16* h0b, bf16* h1b,
       bf16* rh0b, bf16* rh1b, float* z0, float* z1,
       bf16* hs1b, int* bar) {
  int lane = threadIdx.x & 31;
  int wave = global_wave();                  // 0..127
  const bf16* Uh0h = Uh0b + (size_t)2 * H_ * H_;
  const bf16* Wx1h = Wx1b + (size_t)2 * H_ * H_;
  const bf16* Uh1h = Uh1b + (size_t)2 * H_ * H_;
  for (int s = 0; s < S_; ++s) {
    const float* xa_s = xa0 + (size_t)s * B_ * H_;
    // layer 0 r,z:  pre = h0@Uh0[rz]^T + xa0[rz] + b0[rz]
    phase_rz(wave, lane, h0b, Uh0b, nullptr, nullptr, xa_s, SBH_, b0, h0, rh0b, z0);
    grid_barrier(bar);
    // layer 0 h~:   pre = (r*h0)@Uh0[h]^T + xa0[h] + b0[h]
    phase_h(wave, lane, rh0b, Uh0h, nullptr, nullptr,
            xa0 + (size_t)2 * SBH_ + (size_t)s * B_ * H_, b0 + 2 * H_,
            z0, h0, h0b, nullptr);
    grid_barrier(bar);
    // layer 1 r,z:  pre = h0n@Wx1[rz]^T + h1@Uh1[rz]^T + b1[rz]
    phase_rz(wave, lane, h0b, Wx1b, h1b, Uh1b, nullptr, 0, b1, h1, rh1b, z1);
    grid_barrier(bar);
    // layer 1 h~:   pre = h0n@Wx1[h]^T + (r*h1)@Uh1[h]^T + b1[h]; also hs1[s]
    phase_h(wave, lane, h0b, Wx1h, rh1b, Uh1h, nullptr, b1 + 2 * H_,
            z1, h1, h1b, hs1b + (size_t)s * B_ * H_);
    grid_barrier(bar);
  }
}

// ---------------- small helper kernels ----------------
__global__ void k_f2bf(const float* __restrict__ in, bf16* __restrict__ out, int n) {
  int i = blockIdx.x * blockDim.x + threadIdx.x;
  if (i < n) out[i] = (bf16)in[i];
}
__global__ void k_embed(const int* __restrict__ inputs, const float* __restrict__ emb,
                        bf16* __restrict__ xb) {
  int tid = blockIdx.x * blockDim.x + threadIdx.x;  // SB_*E_ threads
  int sb = tid >> 9, e = tid & (E_ - 1);
  xb[tid] = (bf16)emb[(size_t)inputs[sb] * E_ + e];
}
__global__ void k_init_h(const float* __restrict__ hidden, float* h0, float* h1,
                         bf16* h0b, bf16* h1b, int* bar) {
  int i = blockIdx.x * blockDim.x + threadIdx.x;    // B_*H_ threads
  float a = hidden[i], b = hidden[B_ * H_ + i];
  h0[i] = a; h1[i] = b; h0b[i] = (bf16)a; h1b[i] = (bf16)b;
  if (i < 2) bar[i] = 0;                            // reset grid barrier
}
__global__ void k_tail(const float* __restrict__ h0, const float* __restrict__ h1,
                       float* __restrict__ out) {
  int i = blockIdx.x * blockDim.x + threadIdx.x;    // 2*B_*H_ threads
  out[i] = (i < B_ * H_) ? h0[i] : h1[i - B_ * H_];
}

// ---------------- host launcher ----------------
extern "C" void kernel_launch(void* const* d_in, const int* in_sizes, int n_in,
                              void* d_out, int out_size, void* d_ws, size_t ws_size,
                              hipStream_t stream) {
  const int*   inputs = (const int*)  d_in[0];
  const float* hidden = (const float*)d_in[1];
  const float* emb    = (const float*)d_in[2];
  const float* Wx0    = (const float*)d_in[3];
  const float* Uh0    = (const float*)d_in[4];
  const float* b0     = (const float*)d_in[5];
  const float* Wx1    = (const float*)d_in[6];
  const float* Uh1    = (const float*)d_in[7];
  const float* b1     = (const float*)d_in[8];
  const float* Wd     = (const float*)d_in[9];
  const float* bd     = (const float*)d_in[10];
  float* out = (float*)d_out;

  // workspace carve-out (256B aligned)
  char* base = (char*)d_ws; size_t off = 0;
  auto alloc = [&](size_t bytes) -> void* {
    void* p = base + off; off = (off + bytes + 255) & ~(size_t)255; return p;
  };
  bf16*  Wx0b  = (bf16*) alloc((size_t)3 * H_ * E_ * 2);
  bf16*  Uh0b  = (bf16*) alloc((size_t)3 * H_ * H_ * 2);
  bf16*  Wx1b  = (bf16*) alloc((size_t)3 * H_ * H_ * 2);
  bf16*  Uh1b  = (bf16*) alloc((size_t)3 * H_ * H_ * 2);
  bf16*  Wdb   = (bf16*) alloc((size_t)V_ * H_ * 2);
  bf16*  Xembb = (bf16*) alloc((size_t)SB_ * E_ * 2);
  float* xa0   = (float*)alloc((size_t)3 * SBH_ * 4);
  bf16*  hs1b  = (bf16*) alloc(SBH_ * 2);
  float* h0    = (float*)alloc((size_t)B_ * H_ * 4);
  float* h1    = (float*)alloc((size_t)B_ * H_ * 4);
  bf16*  h0b   = (bf16*) alloc((size_t)B_ * H_ * 2);
  bf16*  h1b   = (bf16*) alloc((size_t)B_ * H_ * 2);
  bf16*  rh0b  = (bf16*) alloc((size_t)B_ * H_ * 2);
  bf16*  rh1b  = (bf16*) alloc((size_t)B_ * H_ * 2);
  float* z0    = (float*)alloc((size_t)B_ * H_ * 4);
  float* z1    = (float*)alloc((size_t)B_ * H_ * 4);
  int*   bar   = (int*)  alloc(256);

  // 1) fp32 -> bf16 weight conversion + embedding gather + state/barrier init
  auto cvt = [&](const float* in, bf16* o, size_t n) {
    k_f2bf<<<(int)((n + 255) / 256), 256, 0, stream>>>(in, o, (int)n);
  };
  cvt(Wx0, Wx0b, (size_t)3 * H_ * E_);
  cvt(Uh0, Uh0b, (size_t)3 * H_ * H_);
  cvt(Wx1, Wx1b, (size_t)3 * H_ * H_);
  cvt(Uh1, Uh1b, (size_t)3 * H_ * H_);
  cvt(Wd,  Wdb,  (size_t)V_ * H_);
  k_embed<<<(SB_ * E_) / 256, 256, 0, stream>>>(inputs, emb, Xembb);
  k_init_h<<<(B_ * H_) / 256, 256, 0, stream>>>(hidden, h0, h1, h0b, h1b, bar);

  // 2) xa0[g] = Xemb @ Wx0[g]^T   (M=8192, N=1024, K=512) x3, NT=8 strips
  {
    int waves = (SB_ / 16) * (H_ / 128);
    int blocks = waves * 32 / 256;
    for (int g = 0; g < 3; ++g)
      k_gemm<8><<<blocks, 256, 0, stream>>>(Xembb, E_, Wx0b + (size_t)g * H_ * E_, E_,
                                            nullptr, xa0 + (size_t)g * SBH_, H_,
                                            SB_, H_, E_);
  }

  // 3) whole 128-step scan in ONE persistent kernel (grid barrier between phases)
  k_scan<<<SCAN_BLOCKS, 256, 0, stream>>>(Uh0b, Wx1b, Uh1b, b0, b1, xa0,
                                          h0, h1, h0b, h1b, rh0b, rh1b,
                                          z0, z1, hs1b, bar);

  // 4) logits = hs1 @ Wd^T + bd   (M=8192, N=10000, K=1024)
  {
    // main: 9984 columns (78 strips of 128), NT=8
    int waves = (SB_ / 16) * (9984 / 128);
    int blocks = waves * 32 / 256;
    k_gemm<8><<<blocks, 256, 0, stream>>>(hs1b, H_, Wdb, H_, bd, out, V_,
                                          SB_, 9984, H_);
    // tail: last 16 columns (1 tile per wave)
    int waves_t = (SB_ / 16);
    int blocks_t = waves_t * 32 / 256;
    k_gemm<1><<<blocks_t, 256, 0, stream>>>(hs1b, H_,
                                            Wdb + (size_t)9984 * H_, H_,
                                            bd + 9984, out + 9984, V_,
                                            SB_, 16, H_);
  }
  // 5) final hidden states appended after logits
  k_tail<<<(2 * B_ * H_) / 256, 256, 0, stream>>>(h0, h1, out + (size_t)SB_ * V_);
}
